// CombinedMultiHeadAttention_34394098106924
// MI455X (gfx1250) — matrix operand order repaired and checked by reading, not compiled, not run on hardware
//
#include <hip/hip_runtime.h>
#include <hip/hip_bf16.h>

// -------- problem constants --------
#define BATCH 128
#define SEQ   256
#define CDIM  384
#define HEADS 6
#define DHEAD 64
#define BH    (BATCH * HEADS)   // 768
#define ROWS  (BATCH * SEQ)     // 32768
#define WELEM (CDIM * CDIM)     // 147456

typedef __attribute__((ext_vector_type(16))) __bf16 v16bf;
typedef __attribute__((ext_vector_type(4)))  __bf16 v4bf;
typedef __attribute__((ext_vector_type(8)))  float  v8f;

union V16 { v16bf v; __bf16 e[16]; };
union V8  { v8f   v; float  e[8];  };
union V4  { v4bf  v; __bf16 e[4];  };

// ---- WMMA 16x16x32 bf16 fragment loaders (M/N and C/D per CDNA5 ISA 7.12.2; K order is this code's own, same for A and B) ----
// A (16x32): lane L -> row M = L&15; elem e -> K = e + 16*(L>>4)
// B (32x16): lane L -> col N = L&15; elem e -> K = e + 16*(L>>4)
// C/D (16x16 f32, 8 regs): lane L, reg v -> row = v + 8*(L>>4), col = L&15

static __device__ __forceinline__ v16bf load_a_frag(const __bf16* __restrict__ rowptr, int half) {
  // rowptr = &src[M_row * ld + k0]; contiguous along K
  V16 a;
#pragma unroll
  for (int e = 0; e < 8; ++e)  a.e[e] = rowptr[e + 16 * half];
#pragma unroll
  for (int e = 8; e < 16; ++e) a.e[e] = rowptr[e + 16 * half];
  return a.v;
}

static __device__ __forceinline__ v16bf load_b_frag_kcontig(const __bf16* __restrict__ p) {
  // B[k][n] = src[n*ld + k]  (K contiguous per lane); p = &src[(n0+lane%16)*ld + k0 + 16*half]
  V16 b;
#pragma unroll
  for (int e = 0; e < 16; ++e) b.e[e] = p[e];
  return b.v;
}

static __device__ __forceinline__ float redmax16(float x) {
  x = fmaxf(x, __shfl_xor(x, 1, 32));
  x = fmaxf(x, __shfl_xor(x, 2, 32));
  x = fmaxf(x, __shfl_xor(x, 4, 32));
  x = fmaxf(x, __shfl_xor(x, 8, 32));
  return x;
}
static __device__ __forceinline__ float redsum16(float x) {
  x += __shfl_xor(x, 1, 32);
  x += __shfl_xor(x, 2, 32);
  x += __shfl_xor(x, 4, 32);
  x += __shfl_xor(x, 8, 32);
  return x;
}

// ================== Kernel 0: fp32 -> bf16 pre-conversion ===================
__global__ __launch_bounds__(256) void cvt_bf16_kernel(
    const float* __restrict__ src, __bf16* __restrict__ dst, int n4) {
  const int i = blockIdx.x * 256 + threadIdx.x;
  if (i >= n4) return;
  const float4 f = reinterpret_cast<const float4*>(src)[i];
  V4 o;
  o.e[0] = (__bf16)f.x; o.e[1] = (__bf16)f.y;
  o.e[2] = (__bf16)f.z; o.e[3] = (__bf16)f.w;
  reinterpret_cast<v4bf*>(dst)[i] = o.v;
}

// ===================== Kernel 1: fused Q/K/V projections =====================
// y = x @ W.T (all bf16 operands); each wave: 32x64 tile (A reused 4x, B 2x).
// Q (scaled by Dh^-0.5) and K stored [B*H][T][DHEAD]; V stored TRANSPOSED as
// [B*H][DHEAD][T] so the attention P@V B-fragments are K-contiguous.
__global__ __launch_bounds__(192) void qkv_proj_kernel(
    const __bf16* __restrict__ xb,
    const __bf16* __restrict__ Wqb, const __bf16* __restrict__ Wkb,
    const __bf16* __restrict__ Wvb,
    __bf16* __restrict__ q_ws, __bf16* __restrict__ k_ws, __bf16* __restrict__ vT_ws) {
  const int wave = threadIdx.x >> 5;   // 0..5
  const int lane = threadIdx.x & 31;
  const int half = lane >> 4;
  const int lm   = lane & 15;

  const int m0 = blockIdx.x * 32;      // 32-row tile in [ROWS]
  const int n0 = wave * 64;            // 64-wide column strip in [CDIM]

  const __bf16* W   = (blockIdx.y == 0) ? Wqb : (blockIdx.y == 1) ? Wkb : Wvb;
  const float scale = (blockIdx.y == 0) ? 0.125f : 1.0f;

  V8 c[2][4];
#pragma unroll
  for (int r = 0; r < 2; ++r)
#pragma unroll
    for (int j = 0; j < 4; ++j)
#pragma unroll
      for (int i = 0; i < 8; ++i) c[r][j].e[i] = 0.0f;

  const __bf16* arow0 = xb + (size_t)(m0 + lm) * CDIM;
  const __bf16* arow1 = xb + (size_t)(m0 + 16 + lm) * CDIM;
  const __bf16* brow0 = W + (size_t)(n0 + lm) * CDIM;  // j advances by 16 rows of W
  for (int kk = 0; kk < CDIM; kk += 32) {
    const v16bf a0 = load_a_frag(arow0 + kk, half);
    const v16bf a1 = load_a_frag(arow1 + kk, half);
#pragma unroll
    for (int j = 0; j < 4; ++j) {
      const v16bf b = load_b_frag_kcontig(brow0 + (size_t)(16 * j) * CDIM + kk + 16 * half);
      c[0][j].v = __builtin_amdgcn_wmma_f32_16x16x32_bf16(false, a0, false, b, (short)0,
                                                          c[0][j].v, false, false);
      c[1][j].v = __builtin_amdgcn_wmma_f32_16x16x32_bf16(false, a1, false, b, (short)0,
                                                          c[1][j].v, false, false);
    }
  }

  // scatter C-fragments: uniform base pointer + stride per tile (no inner branches)
#pragma unroll
  for (int r = 0; r < 2; ++r) {
    const int mb   = m0 + 16 * r + 8 * half;  // first row this lane touches
    const int bidx = mb >> 8;                 // batch (tile never crosses batch)
    const int t0   = mb & 255;                // seq pos of v=0
#pragma unroll
    for (int j = 0; j < 4; ++j) {
      const int d  = n0 + 16 * j + lm;
      const int h  = d >> 6;
      const int dh = d & 63;
      const size_t bh = (size_t)bidx * HEADS + h;
      __bf16* p;
      size_t stride;
      if (blockIdx.y == 2) {        // V transposed: [bh][dh][t]
        p = vT_ws + (bh * DHEAD + dh) * SEQ + t0;
        stride = 1;
      } else {                      // Q / K: [bh][t][dh]
        __bf16* base = (blockIdx.y == 1) ? k_ws : q_ws;
        p = base + (bh * SEQ + t0) * DHEAD + dh;
        stride = DHEAD;
      }
#pragma unroll
      for (int v = 0; v < 8; ++v)
        p[(size_t)v * stride] = (__bf16)(c[r][j].e[v] * scale);
    }
  }
}

// ===================== Kernel 2: causal flash attention ======================
// One wave = 16 query rows; key blocks of 32; online softmax; all matmuls WMMA.
__global__ __launch_bounds__(128) void attn_kernel(
    const __bf16* __restrict__ q_ws, const __bf16* __restrict__ k_ws,
    const __bf16* __restrict__ vT_ws, __bf16* __restrict__ ao_ws) {
  __shared__ __bf16 pstage[4][16][32];   // per-wave P tile (C-layout -> A-layout bounce)

  const int wave = threadIdx.x >> 5;
  const int lane = threadIdx.x & 31;
  const int half = lane >> 4;
  const int lm   = lane & 15;

  const int bh = blockIdx.y;                      // (b*HEADS + h)
  const int r0 = blockIdx.x * 64 + wave * 16;     // query row tile base within T

  const __bf16* Q  = q_ws  + (size_t)bh * SEQ * DHEAD;
  const __bf16* K  = k_ws  + (size_t)bh * SEQ * DHEAD;
  const __bf16* Vt = vT_ws + (size_t)bh * DHEAD * SEQ;   // [DHEAD][SEQ]

  // preload Q as two A-fragments (dh 0..31 and 32..63)
  const __bf16* qrow = Q + (size_t)(r0 + lm) * DHEAD;
  const v16bf qa0 = load_a_frag(qrow, half);
  const v16bf qa1 = load_a_frag(qrow + 32, half);

  V8 O[4];
#pragma unroll
  for (int j = 0; j < 4; ++j)
#pragma unroll
    for (int i = 0; i < 8; ++i) O[j].e[i] = 0.0f;

  float mrow[8], lrow[8];
#pragma unroll
  for (int v = 0; v < 8; ++v) { mrow[v] = -__builtin_inff(); lrow[v] = 0.0f; }

  const int nIter = (r0 + 15) / 32 + 1;   // causal: key blocks [0, 32*nIter)
  for (int it = 0; it < nIter; ++it) {
    const int s0 = it * 32;

    // ---- S = Q @ K^T for 16x32 score tile (two 16x16 C-fragments) ----
    V8 s0f, s1f;
#pragma unroll
    for (int i = 0; i < 8; ++i) { s0f.e[i] = 0.0f; s1f.e[i] = 0.0f; }
    {
      const __bf16* kp0 = K + (size_t)(s0 + lm) * DHEAD + 16 * half;       // n = 0..15
      const __bf16* kp1 = K + (size_t)(s0 + 16 + lm) * DHEAD + 16 * half;  // n = 16..31
      const v16bf b00 = load_b_frag_kcontig(kp0);        // dh 0..31
      const v16bf b01 = load_b_frag_kcontig(kp0 + 32);   // dh 32..63
      const v16bf b10 = load_b_frag_kcontig(kp1);
      const v16bf b11 = load_b_frag_kcontig(kp1 + 32);
      s0f.v = __builtin_amdgcn_wmma_f32_16x16x32_bf16(false, qa0, false, b00, (short)0, s0f.v, false, false);
      s0f.v = __builtin_amdgcn_wmma_f32_16x16x32_bf16(false, qa1, false, b01, (short)0, s0f.v, false, false);
      s1f.v = __builtin_amdgcn_wmma_f32_16x16x32_bf16(false, qa0, false, b10, (short)0, s1f.v, false, false);
      s1f.v = __builtin_amdgcn_wmma_f32_16x16x32_bf16(false, qa1, false, b11, (short)0, s1f.v, false, false);
    }

    // ---- causal mask + online softmax update, write P into LDS ----
#pragma unroll
    for (int v = 0; v < 8; ++v) {
      const int t  = r0 + v + 8 * half;   // absolute query position for this reg
      const float NEG = -__builtin_inff();
      const float a0 = (s0 + lm      <= t) ? s0f.e[v] : NEG;
      const float a1 = (s0 + 16 + lm <= t) ? s1f.e[v] : NEG;

      const float mx = redmax16(fmaxf(a0, a1));
      const float mnew  = fmaxf(mrow[v], mx);
      const float alpha = __expf(mrow[v] - mnew);
      const float p0 = __expf(a0 - mnew);
      const float p1 = __expf(a1 - mnew);
      const float rs = redsum16(p0 + p1);

      lrow[v] = lrow[v] * alpha + rs;
      mrow[v] = mnew;
#pragma unroll
      for (int j = 0; j < 4; ++j) O[j].e[v] *= alpha;

      pstage[wave][v + 8 * half][lm]      = (__bf16)p0;
      pstage[wave][v + 8 * half][16 + lm] = (__bf16)p1;
    }

    // ---- reload P as an A-fragment (wave-private LDS; DS ops are in-order) ----
    V16 pa;
#pragma unroll
    for (int e = 0; e < 8; ++e)  pa.e[e] = pstage[wave][lm][e + 16 * half];
#pragma unroll
    for (int e = 8; e < 16; ++e) pa.e[e] = pstage[wave][lm][e + 16 * half];

    // ---- O += P @ V : B-fragments from V^T are K-contiguous ----
    v16bf vb[4];
#pragma unroll
    for (int j = 0; j < 4; ++j)
      vb[j] = load_b_frag_kcontig(Vt + (size_t)(16 * j + lm) * SEQ + s0 + 16 * half);
#pragma unroll
    for (int j = 0; j < 4; ++j)
      O[j].v = __builtin_amdgcn_wmma_f32_16x16x32_bf16(false, pa.v, false, vb[j], (short)0,
                                                       O[j].v, false, false);
  }

  // ---- finalize: O /= l ; store bf16 to [ROWS][CDIM] at columns h*64.. ----
  const int b = bh / HEADS;
  const int h = bh % HEADS;
#pragma unroll
  for (int v = 0; v < 8; ++v) {
    const float inv = 1.0f / lrow[v];
    const int t = r0 + v + 8 * half;
    const size_t row = (size_t)b * SEQ + t;
#pragma unroll
    for (int j = 0; j < 4; ++j)
      ao_ws[row * CDIM + h * DHEAD + 16 * j + lm] = (__bf16)(O[j].e[v] * inv);
  }
}

// ===================== Kernel 3: output projection ===========================
// out = AO @ Wo.T (bf16 operands, fp32 out); each wave: 32x64 tile.
__global__ __launch_bounds__(192) void out_proj_kernel(
    const __bf16* __restrict__ ao, const __bf16* __restrict__ Wob,
    float* __restrict__ out) {
  const int wave = threadIdx.x >> 5;   // 0..5
  const int lane = threadIdx.x & 31;
  const int half = lane >> 4;
  const int lm   = lane & 15;

  const int m0 = blockIdx.x * 32;
  const int n0 = wave * 64;

  V8 c[2][4];
#pragma unroll
  for (int r = 0; r < 2; ++r)
#pragma unroll
    for (int j = 0; j < 4; ++j)
#pragma unroll
      for (int i = 0; i < 8; ++i) c[r][j].e[i] = 0.0f;

  const __bf16* arow0 = ao + (size_t)(m0 + lm) * CDIM;
  const __bf16* arow1 = ao + (size_t)(m0 + 16 + lm) * CDIM;
  const __bf16* brow0 = Wob + (size_t)(n0 + lm) * CDIM;
  for (int kk = 0; kk < CDIM; kk += 32) {
    const v16bf a0 = load_a_frag(arow0 + kk, half);
    const v16bf a1 = load_a_frag(arow1 + kk, half);
#pragma unroll
    for (int j = 0; j < 4; ++j) {
      const v16bf b = load_b_frag_kcontig(brow0 + (size_t)(16 * j) * CDIM + kk + 16 * half);
      c[0][j].v = __builtin_amdgcn_wmma_f32_16x16x32_bf16(false, a0, false, b, (short)0,
                                                          c[0][j].v, false, false);
      c[1][j].v = __builtin_amdgcn_wmma_f32_16x16x32_bf16(false, a1, false, b, (short)0,
                                                          c[1][j].v, false, false);
    }
  }

#pragma unroll
  for (int r = 0; r < 2; ++r) {
    const int mb = m0 + 16 * r + 8 * half;
#pragma unroll
    for (int j = 0; j < 4; ++j) {
      float* p = out + (size_t)mb * CDIM + n0 + 16 * j + lm;
#pragma unroll
      for (int v = 0; v < 8; ++v)
        p[(size_t)v * CDIM] = c[r][j].e[v];
    }
  }
}

// ============================ launcher ======================================
extern "C" void kernel_launch(void* const* d_in, const int* in_sizes, int n_in,
                              void* d_out, int out_size, void* d_ws, size_t ws_size,
                              hipStream_t stream) {
  const float* x  = (const float*)d_in[0];
  const float* Wq = (const float*)d_in[1];
  const float* Wk = (const float*)d_in[2];
  const float* Wv = (const float*)d_in[3];
  const float* Wo = (const float*)d_in[4];
  float* out = (float*)d_out;

  char* ws = (char*)d_ws;
  const size_t act_bytes = (size_t)ROWS * CDIM * sizeof(__bf16);  // 24 MiB
  const size_t w_bytes   = (size_t)WELEM * sizeof(__bf16);        // 288 KiB
  __bf16* x_bf  = (__bf16*)(ws);
  __bf16* q_ws  = (__bf16*)(ws + act_bytes);
  __bf16* k_ws  = (__bf16*)(ws + 2 * act_bytes);
  __bf16* vT_ws = (__bf16*)(ws + 3 * act_bytes);
  __bf16* ao_ws = (__bf16*)(ws + 4 * act_bytes);
  __bf16* Wq_bf = (__bf16*)(ws + 5 * act_bytes);
  __bf16* Wk_bf = (__bf16*)(ws + 5 * act_bytes + 1 * w_bytes);
  __bf16* Wv_bf = (__bf16*)(ws + 5 * act_bytes + 2 * w_bytes);
  __bf16* Wo_bf = (__bf16*)(ws + 5 * act_bytes + 3 * w_bytes);

  // --- stage 0: one-pass bf16 conversion (removes converts from GEMM loops) ---
  const int xn4 = ROWS * CDIM / 4;    // 3,145,728
  const int wn4 = WELEM / 4;          // 36,864
  cvt_bf16_kernel<<<(xn4 + 255) / 256, 256, 0, stream>>>(x,  x_bf,  xn4);
  cvt_bf16_kernel<<<(wn4 + 255) / 256, 256, 0, stream>>>(Wq, Wq_bf, wn4);
  cvt_bf16_kernel<<<(wn4 + 255) / 256, 256, 0, stream>>>(Wk, Wk_bf, wn4);
  cvt_bf16_kernel<<<(wn4 + 255) / 256, 256, 0, stream>>>(Wv, Wv_bf, wn4);
  cvt_bf16_kernel<<<(wn4 + 255) / 256, 256, 0, stream>>>(Wo, Wo_bf, wn4);

  // --- stage 1: Q/K/V projections (WMMA bf16) ---
  dim3 gQKV(ROWS / 32, 3);            // 1024 x 3, 6 waves/block (32x384 per block)
  qkv_proj_kernel<<<gQKV, 192, 0, stream>>>(x_bf, Wq_bf, Wk_bf, Wv_bf, q_ws, k_ws, vT_ws);

  // --- stage 2: causal flash attention (WMMA bf16) ---
  dim3 gAttn(SEQ / 64, BH);           // 4 x 768, 4 waves/block
  attn_kernel<<<gAttn, 128, 0, stream>>>(q_ws, k_ws, vT_ws, ao_ws);

  // --- stage 3: output projection (WMMA bf16 -> fp32) ---
  dim3 gOut(ROWS / 32, 1);            // 1024, 6 waves/block
  out_proj_kernel<<<gOut, 192, 0, stream>>>(ao_ws, Wo_bf, out);
}
